// NerfRenderOccupancy_37486474560116
// MI455X (gfx1250) — compile-verified, matrix operand order checked by reading, not run on hardware
//
#include <hip/hip_runtime.h>
#include <hip/hip_bf16.h>
#include <math.h>

typedef _Float16 v16h __attribute__((ext_vector_type(16)));
typedef _Float16 v8h  __attribute__((ext_vector_type(8)));
typedef float    v8f  __attribute__((ext_vector_type(8)));

#define N_IMAGES 8
#define RAYS 4096
#define SAMP 128
#define GRID_RES 128
#define GRID_N (GRID_RES*GRID_RES*GRID_RES)
#define N_LEVELS 16
#define T_SIZE (1u<<19)
#define BASE_RES 16.0f
#define PLS 1.3819f

// ---------------- WMMA helpers (CDNA5 gfx1250, wave32) ----------------
// Orientation: D(M=neurons, N=samples) = A(Wt: neurons x K) x B(actT: K x samples).
// A and B per-lane layouts share one address pattern over a contiguous K-run:
//   lane&15 selects row(A)/col(B); per lane two contiguous 16B runs
//   (words [4*kh..4*kh+3] and [8+4*kh..11+4*kh], kh = lane>>4) -> 2x ds_load_b128.
// D: lane&15 = sample col, VGPR v = neuron (m0 + v + 8*kh) -> 8 consecutive
//   neurons per lane -> ONE ds_store_b128 after cvt_pk + packed-f16 relu.
// Both column tiles' WMMAs are issued back-to-back so the independent work
// fills the mandatory WMMA->VALU hazard slots (ISA 7.12.1) instead of v_nops.

__device__ inline v8f wmma16x16x32(v16h a, v16h b, v8f c) {
#if defined(__HIP_DEVICE_COMPILE__)
  return __builtin_amdgcn_wmma_f32_16x16x32_f16(
      /*neg_a=*/false, a, /*neg_b=*/false, b,
      /*c_mod=*/(short)0, c, /*reuse_a=*/false, /*reuse_b=*/false);
#else
  (void)a; (void)b; return c;
#endif
}

union frag_cast { uint4 u[2]; v16h h; };

__device__ inline v16h load_frag(const _Float16* __restrict__ base, int stride,
                                 int idx0, int k_off, int lane) {
  int m  = lane & 15;
  int kh = lane >> 4;
  const unsigned* rp = reinterpret_cast<const unsigned*>(base + (idx0 + m) * stride + k_off);
  frag_cast t;
  t.u[0] = *reinterpret_cast<const uint4*>(rp + 4*kh);
  t.u[1] = *reinterpret_cast<const uint4*>(rp + 8 + 4*kh);
  return t.h;
}

// masked variant for narrow weight heads (17-out, 3-out): rows >= nvalid -> 0
__device__ inline v16h load_frag_m(const _Float16* __restrict__ base, int stride,
                                   int idx0, int nvalid, int k_off, int lane) {
  int m  = lane & 15;
  int kh = lane >> 4;
  frag_cast t;
  if (idx0 + m < nvalid) {
    const unsigned* rp = reinterpret_cast<const unsigned*>(base + (idx0 + m) * stride + k_off);
    t.u[0] = *reinterpret_cast<const uint4*>(rp + 4*kh);
    t.u[1] = *reinterpret_cast<const uint4*>(rp + 8 + 4*kh);
  } else {
    t.u[0] = make_uint4(0u,0u,0u,0u);
    t.u[1] = make_uint4(0u,0u,0u,0u);
  }
  return t.h;
}

// D-fragment store: cvt to f16 first (4x cvt_pk), relu in packed f16
// (4x v_pk_max_num_f16), then one ds_store_b128.
__device__ inline void store_dT_relu(_Float16* __restrict__ base, int stride,
                                     int m0, int c0, v8f c, int lane) {
  int nn = lane & 15;
  int kh = lane >> 4;
  v8h hv;
#pragma unroll
  for (int v = 0; v < 8; ++v) hv[v] = (_Float16)c[v];
  v8h z = {};
#if defined(__HIP_DEVICE_COMPILE__)
  hv = __builtin_elementwise_max(hv, z);
#else
  for (int v = 0; v < 8; ++v) hv[v] = hv[v] > z[v] ? hv[v] : z[v];
#endif
  union { v8h h; uint4 u; } p;
  p.h = hv;
  *reinterpret_cast<uint4*>(base + (c0 + nn)*stride + m0 + 8*kh) = p.u;
}

// ---------------- density-grid mean reduction ----------------

__global__ void ws_init_kernel(float* s) {
  if (threadIdx.x == 0 && blockIdx.x == 0) s[0] = 0.0f;
}

__global__ void grid_sum_kernel(const float* __restrict__ g, float* __restrict__ sum, int n) {
  __shared__ float red[256];
  int i = blockIdx.x * 256 + threadIdx.x;
  red[threadIdx.x] = (i < n) ? g[i] : 0.0f;
  __syncthreads();
  for (int s = 128; s > 0; s >>= 1) {
    if ((int)threadIdx.x < s) red[threadIdx.x] += red[threadIdx.x + s];
    __syncthreads();
  }
  if (threadIdx.x == 0) atomicAdd(sum, red[0]);
}

// ---------------- fused NeRF render kernel ----------------
// 1 block = 1 ray; 128 threads = 4 wave32; thread = sample.
// Samples are N-columns; each wave owns columns [32*wave, +32) -> two 16-col tiles.

__global__ __launch_bounds__(128)
void nerf_render_kernel(const int* __restrict__ nIdx, const int* __restrict__ hPix,
                        const int* __restrict__ wPix,
                        const float* __restrict__ intr,   // (8,3,3)
                        const float* __restrict__ extr,   // (8,4,4)
                        const float* __restrict__ bds,    // (8,2)
                        const float* __restrict__ tnoise, // (R,S)
                        const float* __restrict__ table,  // (16,T,2)
                        const float* __restrict__ sw1,    // (32,64)
                        const float* __restrict__ sw2,    // (64,17)
                        const float* __restrict__ rw1,    // (32,64)
                        const float* __restrict__ rw2,    // (64,64)
                        const float* __restrict__ rw3,    // (64,3)
                        const float* __restrict__ grid,   // (128^3)
                        const float* __restrict__ bg,     // (3)
                        const float* __restrict__ gsum,   // scalar
                        float* __restrict__ out_img,      // (R,3)
                        float* __restrict__ out_depth) {  // (R)
  // transposed activations actT[sample][feature] (f16, 16B-aligned rows)
  __shared__ __align__(16) _Float16 sBufT[SAMP*64];   // enc (feats 0..31) -> rgb hidden2
  __shared__ __align__(16) _Float16 sActT[SAMP*64];   // sigma hidden / rgb hidden1
  __shared__ __align__(16) _Float16 sRgbInT[SAMP*32]; // [SH(16) | geo(16)]
  // weights TRANSPOSED: WT[n*K + k] (row = output neuron)
  __shared__ __align__(16) _Float16 sW1T [64*32];
  __shared__ __align__(16) _Float16 sW2T [17*64];
  __shared__ __align__(16) _Float16 sRW1T[64*32];
  __shared__ __align__(16) _Float16 sRW2T[64*64];
  __shared__ __align__(16) _Float16 sRW3T[3*64];
  __shared__ float sSigma[SAMP];  // phase1: occ flag; MLP: masked relu(sigma)
  __shared__ float sT[SAMP];
  __shared__ float sRGB[SAMP*3];

  const int r    = blockIdx.x;
  const int tid  = threadIdx.x;
  const int lane = tid & 31;
  const int wave = tid >> 5;

  // ---- cooperative transposed weight load (f32 -> f16) ----
  for (int i = tid; i < 64*32; i += 128) {
    int n = i >> 5, k = i & 31;
    sW1T [i] = (_Float16)sw1[k*64 + n];
    sRW1T[i] = (_Float16)rw1[k*64 + n];
  }
  for (int i = tid; i < 17*64; i += 128) {
    int n = i >> 6, k = i & 63;
    sW2T[i] = (_Float16)sw2[k*17 + n];
  }
  for (int i = tid; i < 64*64; i += 128) {
    int n = i >> 6, k = i & 63;
    sRW2T[i] = (_Float16)rw2[k*64 + n];
  }
  for (int i = tid; i < 3*64; i += 128) {
    int n = i >> 6, k = i & 63;
    sRW3T[i] = (_Float16)rw3[k*3 + n];
  }

  // ---- per-sample ray setup ----
  const int ni = nIdx[r];
  const float fxx = intr[ni*9 + 0], fyy = intr[ni*9 + 4];
  const float cx = intr[ni*9 + 2], cy = intr[ni*9 + 5];
  const float wf = (float)wPix[r], hf = (float)hPix[r];
  float dcx = (wf + 0.5f - cx) / fxx;
  float dcy = (hf + 0.5f - cy) / fyy;
  const float* E = extr + ni*16;
  float dx = E[0]*dcx + E[1]*dcy + E[2];
  float dy = E[4]*dcx + E[5]*dcy + E[6];
  float dz = E[8]*dcx + E[9]*dcy + E[10];
  float rn = rsqrtf(dx*dx + dy*dy + dz*dz);
  dx *= rn; dy *= rn; dz *= rn;
  const float ox = E[3], oy = E[7], oz = E[11];
  const float nearp = bds[ni*2 + 0], farp = bds[ni*2 + 1];

  const float t = nearp + (farp - nearp) * ((float)tid + tnoise[r*SAMP + tid]) * (1.0f/(float)SAMP);
  sT[tid] = t;

  float px = fminf(fmaxf(ox + dx*t, -1.0f), 1.0f);
  float py = fminf(fmaxf(oy + dy*t, -1.0f), 1.0f);
  float pz = fminf(fmaxf(oz + dz*t, -1.0f), 1.0f);

  // occupancy flag stashed in sSigma (overwritten with masked sigma in MLP)
  {
    int gx = min(max((int)((px + 1.0f) * 0.5f * GRID_RES), 0), GRID_RES-1);
    int gy = min(max((int)((py + 1.0f) * 0.5f * GRID_RES), 0), GRID_RES-1);
    int gz = min(max((int)((pz + 1.0f) * 0.5f * GRID_RES), 0), GRID_RES-1);
    float mean = gsum[0] * (1.0f / (float)GRID_N);
    sSigma[tid] = (grid[(gx*GRID_RES + gy)*GRID_RES + gz] > mean) ? 1.0f : 0.0f;
  }

  // ---- hash encoding (64MB table lives in the 192MB L2; b64 gathers) ----
  {
    const float u0 = (px + 1.0f) * 0.5f;
    const float u1 = (py + 1.0f) * 0.5f;
    const float u2 = (pz + 1.0f) * 0.5f;
    float growth = 1.0f;
    for (int l = 0; l < N_LEVELS; ++l) {
      float resl = floorf(BASE_RES * growth);
      growth *= PLS;
      float qx = u0 * resl, qy = u1 * resl, qz = u2 * resl;
      float f0x = floorf(qx), f0y = floorf(qy), f0z = floorf(qz);
      float frx = qx - f0x, fry = qy - f0y, frz = qz - f0z;
      unsigned ix = (unsigned)f0x, iy = (unsigned)f0y, iz = (unsigned)f0z;
      float a0 = 0.0f, a1 = 0.0f;
#pragma unroll
      for (int c = 0; c < 8; ++c) {
        unsigned ccx = ix + (c & 1);
        unsigned ccy = iy + ((c >> 1) & 1);
        unsigned ccz = iz + (c >> 2);
        unsigned hh = (ccx * 1u) ^ (ccy * 2654435761u) ^ (ccz * 805459861u);
        hh &= (T_SIZE - 1u);
        float wcx = (c & 1)        ? frx : 1.0f - frx;
        float wcy = ((c >> 1) & 1) ? fry : 1.0f - fry;
        float wcz = (c >> 2)       ? frz : 1.0f - frz;
        float wgt = wcx * wcy * wcz;
        const float2 e = *reinterpret_cast<const float2*>(table + ((size_t)l * T_SIZE + hh) * 2);
        a0 = fmaf(e.x, wgt, a0);
        a1 = fmaf(e.y, wgt, a1);
      }
      sBufT[tid*64 + 2*l    ] = (_Float16)a0;
      sBufT[tid*64 + 2*l + 1] = (_Float16)a1;
    }
  }

  // ---- SH deg 4 (16 coeffs) into rgb-input feats 0..15 ----
  {
    float xx = dx*dx, yy = dy*dy, zz = dz*dz;
    _Float16* sh = sRgbInT + tid*32;
    sh[0]  = (_Float16)(0.28209479177387814f);
    sh[1]  = (_Float16)(-0.48860251190291987f * dy);
    sh[2]  = (_Float16)( 0.48860251190291987f * dz);
    sh[3]  = (_Float16)(-0.48860251190291987f * dx);
    sh[4]  = (_Float16)( 1.0925484305920792f * dx * dy);
    sh[5]  = (_Float16)(-1.0925484305920792f * dy * dz);
    sh[6]  = (_Float16)( 0.94617469575756f * zz - 0.31539156525252005f);
    sh[7]  = (_Float16)(-1.0925484305920792f * dx * dz);
    sh[8]  = (_Float16)( 0.5462742152960396f * (xx - yy));
    sh[9]  = (_Float16)(-0.5900435899266435f * dy * (3.0f*xx - yy));
    sh[10] = (_Float16)( 2.890611442640554f * dx * dy * dz);
    sh[11] = (_Float16)( 0.4570457994644657f * dy * (1.0f - 5.0f*zz));
    sh[12] = (_Float16)( 0.3731763325901154f * dz * (5.0f*zz - 3.0f));
    sh[13] = (_Float16)( 0.4570457994644657f * dx * (1.0f - 5.0f*zz));
    sh[14] = (_Float16)( 1.445305721320277f * dz * (xx - yy));
    sh[15] = (_Float16)(-0.5900435899266435f * dx * (xx - 3.0f*yy));
  }

  __syncthreads();

  // ---- MLP phase (Yt = Wt x Xt): wave owns sample columns [32*wave, +32) ----
  {
    const int c0 = wave*32;
    const int c1 = wave*32 + 16;
    const int nn = lane & 15;
    const int kh = lane >> 4;

    // sigma layer1: W1T(64x32) x encT(32 x cols) -> relu -> sActT
    {
      v16h a[4];
#pragma unroll
      for (int mt = 0; mt < 4; ++mt) a[mt] = load_frag(sW1T, 32, mt*16, 0, lane);
      v16h bA = load_frag(sBufT, 64, c0, 0, lane);   // enc feats 0..31
      v16h bB = load_frag(sBufT, 64, c1, 0, lane);
#pragma unroll
      for (int mt = 0; mt < 4; ++mt) {
        v8f cA = {}; cA = wmma16x16x32(a[mt], bA, cA);
        v8f cB = {}; cB = wmma16x16x32(a[mt], bB, cB);
        store_dT_relu(sActT, 64, mt*16, c0, cA, lane);
        store_dT_relu(sActT, 64, mt*16, c1, cB, lane);
      }
    }

    // sigma layer2: W2T(17x64) x act(64 x cols) -> row0 sigma, rows1..16 geo
    {
      v16h a00 = load_frag_m(sW2T, 64, 0,  17, 0,  lane);
      v16h a01 = load_frag_m(sW2T, 64, 0,  17, 32, lane);
      v16h a10 = load_frag_m(sW2T, 64, 16, 17, 0,  lane);
      v16h a11 = load_frag_m(sW2T, 64, 16, 17, 32, lane);
      v16h b0A = load_frag(sActT, 64, c0, 0,  lane);
      v16h b1A = load_frag(sActT, 64, c0, 32, lane);
      v16h b0B = load_frag(sActT, 64, c1, 0,  lane);
      v16h b1B = load_frag(sActT, 64, c1, 32, lane);
      // neurons 0..15 for both column tiles (interleaved)
      v8f cA = {}; cA = wmma16x16x32(a00, b0A, cA);
      v8f cB = {}; cB = wmma16x16x32(a00, b0B, cB);
      cA = wmma16x16x32(a01, b1A, cA);
      cB = wmma16x16x32(a01, b1B, cB);
      // neuron 16 (rest masked to zero)
      v8f dA = {}; dA = wmma16x16x32(a10, b0A, dA);
      v8f dB = {}; dB = wmma16x16x32(a10, b0B, dB);
      dA = wmma16x16x32(a11, b1A, dA);
      dB = wmma16x16x32(a11, b1B, dB);
      int sA = c0 + nn, sB = c1 + nn;
#pragma unroll
      for (int v = 0; v < 8; ++v) {
        int neuron = v + 8*kh;
        if (neuron == 0) {
          sSigma[sA] = (sSigma[sA] != 0.0f) ? fmaxf(cA[v], 0.0f) : 0.0f;
          sSigma[sB] = (sSigma[sB] != 0.0f) ? fmaxf(cB[v], 0.0f) : 0.0f;
        } else {
          sRgbInT[sA*32 + 16 + (neuron - 1)] = (_Float16)cA[v];  // geo: no relu
          sRgbInT[sB*32 + 16 + (neuron - 1)] = (_Float16)cB[v];
        }
      }
      if (kh == 0) {
        sRgbInT[sA*32 + 31] = (_Float16)dA[0];                   // geo 15 (neuron 16)
        sRgbInT[sB*32 + 31] = (_Float16)dB[0];
      }
    }

    // rgb layer1: RW1T(64x32) x [SH|geo](32 x cols) -> relu -> sActT (reuse)
    {
      v16h a[4];
#pragma unroll
      for (int mt = 0; mt < 4; ++mt) a[mt] = load_frag(sRW1T, 32, mt*16, 0, lane);
      v16h bA = load_frag(sRgbInT, 32, c0, 0, lane);
      v16h bB = load_frag(sRgbInT, 32, c1, 0, lane);
#pragma unroll
      for (int mt = 0; mt < 4; ++mt) {
        v8f cA = {}; cA = wmma16x16x32(a[mt], bA, cA);
        v8f cB = {}; cB = wmma16x16x32(a[mt], bB, cB);
        store_dT_relu(sActT, 64, mt*16, c0, cA, lane);
        store_dT_relu(sActT, 64, mt*16, c1, cB, lane);
      }
    }

    // rgb layer2: RW2T(64x64) x act(64 x cols) -> relu -> sBufT (enc dead)
    {
      v16h a0[4], a1[4];
#pragma unroll
      for (int mt = 0; mt < 4; ++mt) {
        a0[mt] = load_frag(sRW2T, 64, mt*16, 0,  lane);
        a1[mt] = load_frag(sRW2T, 64, mt*16, 32, lane);
      }
      v16h b0A = load_frag(sActT, 64, c0, 0,  lane);
      v16h b1A = load_frag(sActT, 64, c0, 32, lane);
      v16h b0B = load_frag(sActT, 64, c1, 0,  lane);
      v16h b1B = load_frag(sActT, 64, c1, 32, lane);
#pragma unroll
      for (int mt = 0; mt < 4; ++mt) {
        v8f cA = {}; cA = wmma16x16x32(a0[mt], b0A, cA);
        v8f cB = {}; cB = wmma16x16x32(a0[mt], b0B, cB);
        cA = wmma16x16x32(a1[mt], b1A, cA);
        cB = wmma16x16x32(a1[mt], b1B, cB);
        store_dT_relu(sBufT, 64, mt*16, c0, cA, lane);
        store_dT_relu(sBufT, 64, mt*16, c1, cB, lane);
      }
    }

    // rgb layer3: RW3T(3x64) x hidden2(64 x cols) -> sigmoid -> sRGB
    {
      v16h a0 = load_frag_m(sRW3T, 64, 0, 3, 0,  lane);
      v16h a1 = load_frag_m(sRW3T, 64, 0, 3, 32, lane);
      v16h b0A = load_frag(sBufT, 64, c0, 0,  lane);
      v16h b1A = load_frag(sBufT, 64, c0, 32, lane);
      v16h b0B = load_frag(sBufT, 64, c1, 0,  lane);
      v16h b1B = load_frag(sBufT, 64, c1, 32, lane);
      v8f cA = {}; cA = wmma16x16x32(a0, b0A, cA);
      v8f cB = {}; cB = wmma16x16x32(a0, b0B, cB);
      cA = wmma16x16x32(a1, b1A, cA);
      cB = wmma16x16x32(a1, b1B, cB);
      if (kh == 0) {
        int sA = c0 + nn, sB = c1 + nn;
#pragma unroll
        for (int v = 0; v < 3; ++v) {
          sRGB[sA*3 + v] = 1.0f / (1.0f + __expf(-cA[v]));
          sRGB[sB*3 + v] = 1.0f / (1.0f + __expf(-cB[v]));
        }
      }
    }
  }

  __syncthreads();

  // ---- volume rendering: parallel scan over 128 samples ----
  float* scan = reinterpret_cast<float*>(sActT);  // f16 buffers dead; reuse as f32 scratch
  const float ts = sT[tid];
  const float dl = (tid < SAMP-1) ? (sT[tid+1] - ts) : (farp - ts);
  const float sd = sSigma[tid] * dl;

  scan[tid] = sd;
  __syncthreads();
#pragma unroll
  for (int off = 1; off < SAMP; off <<= 1) {
    float mine = scan[tid];
    float add  = (tid >= off) ? scan[tid - off] : 0.0f;
    __syncthreads();
    scan[tid] = mine + add;
    __syncthreads();
  }
  const float cuminc = scan[tid];
  const float Tr  = __expf(-(cuminc - sd));     // exclusive prefix
  const float wgt = (1.0f - __expf(-sd)) * Tr;

  // 5-way tree reduction (im0,im1,im2,depth,acc) in reused LDS
  float* red = reinterpret_cast<float*>(sActT);
  __syncthreads();
  red[0*SAMP + tid] = wgt * sRGB[tid*3 + 0];
  red[1*SAMP + tid] = wgt * sRGB[tid*3 + 1];
  red[2*SAMP + tid] = wgt * sRGB[tid*3 + 2];
  red[3*SAMP + tid] = wgt * ts;
  red[4*SAMP + tid] = wgt;
  __syncthreads();
#pragma unroll
  for (int off = 64; off > 0; off >>= 1) {
    if (tid < off) {
#pragma unroll
      for (int q = 0; q < 5; ++q)
        red[q*SAMP + tid] += red[q*SAMP + tid + off];
    }
    __syncthreads();
  }
  if (tid == 0) {
    float acc = red[4*SAMP];
    float bgw = 1.0f - acc;
    out_img[r*3 + 0] = red[0*SAMP] + bgw * bg[0];
    out_img[r*3 + 1] = red[1*SAMP] + bgw * bg[1];
    out_img[r*3 + 2] = red[2*SAMP] + bgw * bg[2];
    out_depth[r]     = red[3*SAMP];
  }
}

// ---------------- launch ----------------

extern "C" void kernel_launch(void* const* d_in, const int* in_sizes, int n_in,
                              void* d_out, int out_size, void* d_ws, size_t ws_size,
                              hipStream_t stream) {
  (void)in_sizes; (void)n_in; (void)out_size; (void)ws_size;
  const int*   nI   = (const int*)d_in[0];
  const int*   hI   = (const int*)d_in[1];
  const int*   wI   = (const int*)d_in[2];
  const float* intr = (const float*)d_in[3];
  const float* extr = (const float*)d_in[4];
  const float* bdsP = (const float*)d_in[5];
  const float* tns  = (const float*)d_in[6];
  const float* tbl  = (const float*)d_in[7];
  const float* sw1  = (const float*)d_in[8];
  const float* sw2  = (const float*)d_in[9];
  const float* rw1  = (const float*)d_in[10];
  const float* rw2  = (const float*)d_in[11];
  const float* rw3  = (const float*)d_in[12];
  const float* grd  = (const float*)d_in[13];
  const float* bg   = (const float*)d_in[14];

  float* out       = (float*)d_out;
  float* out_img   = out;              // (R,3)
  float* out_depth = out + RAYS*3;     // (R)
  float* gsum      = (float*)d_ws;

  ws_init_kernel<<<1, 32, 0, stream>>>(gsum);
  grid_sum_kernel<<<(GRID_N + 255)/256, 256, 0, stream>>>(grd, gsum, GRID_N);
  nerf_render_kernel<<<RAYS, 128, 0, stream>>>(
      nI, hI, wI, intr, extr, bdsP, tns, tbl,
      sw1, sw2, rw1, rw2, rw3, grd, bg, gsum, out_img, out_depth);
}
